// GraphTransformer_2619930050605
// MI455X (gfx1250) — compile-verified, matrix operand order checked by reading, not run on hardware
//
#include <hip/hip_runtime.h>

// ---------------------------------------------------------------------------
// Graph Transformer (2x PyG TransformerConv + mean-pool + linear) for gfx1250.
// Dense projections: v_wmma_f32_16x16x32_bf16, A staged in LDS via
// global_load_async_to_lds_b128 (ASYNCcnt) and read back with ds_load_b128.
// Edge softmax/aggregation: native u32/f32 global atomics, 2 edge passes.
// ---------------------------------------------------------------------------

#define NODE_ROW 416   // fused row: q(128) | k(128) | v(128) | skip(32)
#define HD 4
#define CD 32
#define NTILES 26      // 416 / 16

typedef __bf16 bf16_t;
typedef __attribute__((ext_vector_type(16))) __bf16 v16bf;
typedef __attribute__((ext_vector_type(8)))  __bf16 v8bf;
typedef __attribute__((ext_vector_type(8)))  float  v8f;

// ------------------------- small helpers -----------------------------------

// Monotonic order-preserving key for f32 so segment-max can use native
// global_atomic_max_u32 and be initialized with memset(0).
__device__ __forceinline__ unsigned int f32_order_key(float f) {
    unsigned int b = __float_as_uint(f);
    return (b & 0x80000000u) ? ~b : (b | 0x80000000u);
}
__device__ __forceinline__ float key_to_f32(unsigned int k) {
    unsigned int b = (k & 0x80000000u) ? (k & 0x7FFFFFFFu) : ~k;
    return __uint_as_float(b);
}

__device__ __forceinline__ void async_wait_zero() {
#if __has_builtin(__builtin_amdgcn_s_wait_asynccnt)
    __builtin_amdgcn_s_wait_asynccnt(0);
#else
    asm volatile("s_wait_asynccnt 0x0" ::: "memory");
#endif
}

// ------------------------- prep kernels ------------------------------------

__global__ void gt_cast_bf16_kernel(const float* __restrict__ in,
                                    bf16_t* __restrict__ out, int n) {
    int t = blockIdx.x * blockDim.x + threadIdx.x;
    if (t < n) out[t] = (bf16_t)in[t];
}

// Concatenate biases: [bq(128) | bk(128) | bv(128) | bs(32)]
__global__ void gt_pack_bias_kernel(const float* __restrict__ bq,
                                    const float* __restrict__ bk,
                                    const float* __restrict__ bv,
                                    const float* __restrict__ bs,
                                    float* __restrict__ out) {
    int t = blockIdx.x * blockDim.x + threadIdx.x;
    if (t >= NODE_ROW) return;
    float v;
    if      (t < 128) v = bq[t];
    else if (t < 256) v = bk[t - 128];
    else if (t < 384) v = bv[t - 256];
    else              v = bs[t - 384];
    out[t] = v;
}

// Pack fused weight matrix [K, 416] into WMMA B-fragment order (bf16).
// 16-bit operand layout (CDNA5 ISA): lane L<16 holds column N=L
// with K in {k0..k0+7, k0+16..k0+23}; lane L>=16 holds N=L-16 with the
// other K-halves.  One thread per (kstep, nTile, lane).
__global__ void gt_pack_w_kernel(const float* __restrict__ Wq,
                                 const float* __restrict__ Wk,
                                 const float* __restrict__ Wv,
                                 const float* __restrict__ Ws,
                                 bf16_t* __restrict__ out, int K) {
    int t = blockIdx.x * blockDim.x + threadIdx.x;
    int total = (K >> 5) * NTILES * 32;
    if (t >= total) return;
    int lane = t & 31;
    int tN   = (t >> 5) % NTILES;
    int ks   = (t >> 5) / NTILES;
    int n    = tN * 16 + (lane & 15);
    int kh   = lane >> 4;
    bf16_t* o = out + (size_t)t * 16;
#pragma unroll
    for (int i = 0; i < 16; ++i) {
        int k = ks * 32 + kh * 8 + ((i < 8) ? i : (i - 8) + 16);
        float v;
        if      (n < 128) v = Wq[(size_t)k * 128 + n];
        else if (n < 256) v = Wk[(size_t)k * 128 + (n - 128)];
        else if (n < 384) v = Wv[(size_t)k * 128 + (n - 256)];
        else              v = Ws[(size_t)k * 32  + (n - 384)];
        o[i] = (bf16_t)v;
    }
}

// ------------------------- WMMA GEMM ---------------------------------------
// D[M,416] = A[M,K](bf16, row-major) x Bpacked[K,416](bf16 fragments) + bias.
// Block = 4 waves. The 16-row A stripe is staged once into LDS with
// global_load_async_to_lds_b128 and shared by all waves; each wave computes
// TWO adjacent 16x16 N-tiles per A fragment (8 tiles / block).
template <int K>
__global__ void __launch_bounds__(128)
gt_wmma_gemm_kernel(const bf16_t* __restrict__ A,
                    const bf16_t* __restrict__ Bp,
                    const float* __restrict__ bias,
                    float* __restrict__ D) {
    constexpr int LDSTRIDE = K * 2 + 16;          // bytes/row, 16B-pad vs banks
    __shared__ __align__(16) char smem[16 * LDSTRIDE];

    const int lane = threadIdx.x;                 // 0..31 (wave32)
    const int wave = threadIdx.y;                 // 0..3
    const int tid  = wave * 32 + lane;
    const int tM   = blockIdx.x;

    // ---- stage A stripe (16 rows x K bf16) into LDS, ASYNCcnt-tracked ----
    constexpr int CHUNKS_PER_ROW = K / 8;         // 16B chunks per row
    constexpr int TOTAL_CHUNKS   = 16 * CHUNKS_PER_ROW;
    for (int c = tid; c < TOTAL_CHUNKS; c += 128) {
        int r   = c / CHUNKS_PER_ROW;
        int col = c - r * CHUNKS_PER_ROW;
        const bf16_t* gsrc = A + ((size_t)tM * 16 + r) * K + col * 8;
        unsigned lds_addr = (unsigned)(uintptr_t)(&smem[r * LDSTRIDE + col * 16]);
        unsigned long long gaddr = (unsigned long long)(uintptr_t)gsrc;
        asm volatile("global_load_async_to_lds_b128 %0, %1, off"
                     :: "v"(lds_addr), "v"(gaddr) : "memory");
    }
    async_wait_zero();
    __syncthreads();

    const int tN0 = blockIdx.y * 8 + wave * 2;    // wave-uniform
    if (tN0 >= NTILES) return;                    // after barrier: safe
    const bool has1 = (tN0 + 1) < NTILES;         // wave-uniform

    const int l15 = lane & 15;
    const int kh  = lane >> 4;
    const int n0  = tN0 * 16 + l15;

    v8f acc0, acc1;
    const float bv0 = bias[n0];
    const float bv1 = has1 ? bias[n0 + 16] : 0.f;
#pragma unroll
    for (int j = 0; j < 8; ++j) { acc0[j] = bv0; acc1[j] = bv1; }

    const char* aptr = smem + l15 * LDSTRIDE + kh * 16;
    const bf16_t* Bq = Bp + ((size_t)tN0 * 32 + lane) * 16;
    constexpr size_t BSTRIDE = (size_t)NTILES * 32 * 16;   // per-kstep

#pragma unroll
    for (int ks = 0; ks < K / 32; ++ks) {
        v8bf alo = *(const v8bf*)(aptr);          // ds_load_b128
        v8bf ahi = *(const v8bf*)(aptr + 32);     // ds_load_b128
        v16bf a = __builtin_shufflevector(alo, ahi,
                    0, 1, 2, 3, 4, 5, 6, 7, 8, 9, 10, 11, 12, 13, 14, 15);
        v16bf b0 = *(const v16bf*)(Bq);
        acc0 = __builtin_amdgcn_wmma_f32_16x16x32_bf16(
                   false, a, false, b0, (short)0, acc0, false, false);
        if (has1) {
            v16bf b1 = *(const v16bf*)(Bq + 32 * 16);      // adjacent tile
            acc1 = __builtin_amdgcn_wmma_f32_16x16x32_bf16(
                       false, a, false, b1, (short)0, acc1, false, false);
        }
        aptr += 64;
        Bq   += BSTRIDE;
    }

    // C/D layout: VGPR j, lanes 0-15 -> M=j, lanes 16-31 -> M=8+j; N=lane&15.
    float* Drow = D + ((size_t)tM * 16 + kh * 8) * NODE_ROW + n0;
#pragma unroll
    for (int j = 0; j < 8; ++j) Drow[(size_t)j * NODE_ROW] = acc0[j];
    if (has1) {
#pragma unroll
        for (int j = 0; j < 8; ++j) Drow[(size_t)j * NODE_ROW + 16] = acc1[j];
    }
}

// ------------------------- edge passes -------------------------------------

// Pass 1: alpha[e,h] = <q[dst,h,:], k[src,h,:]>/sqrt(C); segment-max via key.
__global__ void gt_edge_alpha_kernel(const float* __restrict__ qkvs,
                                     const long long* __restrict__ ei,
                                     float* __restrict__ alpha,
                                     unsigned int* __restrict__ amax_key,
                                     int E) {
    int t = blockIdx.x * blockDim.x + threadIdx.x;
    if (t >= E * HD) return;
    int e = t >> 2, h = t & 3;
    int src = (int)ei[e];
    int dst = (int)ei[E + e];
    const float4* qp = (const float4*)(qkvs + (size_t)dst * NODE_ROW + h * CD);
    const float4* kp = (const float4*)(qkvs + (size_t)src * NODE_ROW + 128 + h * CD);
    // warm the V row we re-read in pass 2
    __builtin_prefetch(qkvs + (size_t)src * NODE_ROW + 256 + h * CD, 0, 1);
    float s = 0.f;
#pragma unroll
    for (int i = 0; i < 8; ++i) {
        float4 q4 = qp[i], k4 = kp[i];
        s += q4.x * k4.x + q4.y * k4.y + q4.z * k4.z + q4.w * k4.w;
    }
    s *= 0.17677669529663687f;                    // 1/sqrt(32)
    alpha[t] = s;
    atomicMax(&amax_key[dst * HD + h], f32_order_key(s));
}

// Pass 2 (fused exp+scatter, normalization deferred to node-finish):
//   a = exp(alpha - amax[dst]);  denom[dst,h] += a;  out[dst,h,:] += a*v[src,h,:]
__global__ void gt_edge_exp_scatter_kernel(const float* __restrict__ qkvs,
                                           const float* __restrict__ alpha,
                                           const unsigned int* __restrict__ amax_key,
                                           float* __restrict__ denom,
                                           float* __restrict__ outacc,
                                           const long long* __restrict__ ei, int E) {
    int t = blockIdx.x * blockDim.x + threadIdx.x;
    if (t >= E * HD) return;
    int e = t >> 2, h = t & 3;
    int src = (int)ei[e];
    int dst = (int)ei[E + e];
    float m = key_to_f32(amax_key[dst * HD + h]);
    float a = __expf(alpha[t] - m);
    atomicAdd(&denom[dst * HD + h], a);
    const float4* vp = (const float4*)(qkvs + (size_t)src * NODE_ROW + 256 + h * CD);
    float* op = outacc + ((size_t)dst * HD + h) * CD;
#pragma unroll
    for (int i = 0; i < 8; ++i) {
        float4 v4 = vp[i];
        atomicAdd(op + i * 4 + 0, v4.x * a);
        atomicAdd(op + i * 4 + 1, v4.y * a);
        atomicAdd(op + i * 4 + 2, v4.z * a);
        atomicAdd(op + i * 4 + 3, v4.w * a);
    }
}

// out = mean over heads of (outacc/denom) + skip.
// mode 0: relu -> bf16 (layer1 hidden), mode 1: f32 (layer2 output)
__global__ void gt_node_finish_kernel(const float* __restrict__ outacc,
                                      const float* __restrict__ denom,
                                      const float* __restrict__ qkvs,
                                      bf16_t* __restrict__ out_bf,
                                      float* __restrict__ out_f,
                                      int mode, int N) {
    int t = blockIdx.x * blockDim.x + threadIdx.x;
    if (t >= N * CD) return;
    int nid = t >> 5, c = t & 31;
    const float* oa = outacc + (size_t)nid * (HD * CD);
    const float* dn = denom + (size_t)nid * HD;
    float s = 0.f;
#pragma unroll
    for (int h = 0; h < HD; ++h)
        s += oa[h * CD + c] / (dn[h] + 1e-16f);
    s *= 0.25f;                                   // head mean
    s += qkvs[(size_t)nid * NODE_ROW + 384 + c];  // root skip (+bias, in GEMM)
    if (mode == 0) {
        s = fmaxf(s, 0.f);
        out_bf[t] = (bf16_t)s;
    } else {
        out_f[t] = s;
    }
}

// ------------------------- pooling + head ----------------------------------

__global__ void gt_pool_kernel(const float* __restrict__ h,
                               const long long* __restrict__ batch,
                               float* __restrict__ sums,
                               float* __restrict__ cnts, int N) {
    int t = blockIdx.x * blockDim.x + threadIdx.x;
    if (t >= N * CD) return;
    int nid = t >> 5, c = t & 31;
    int g = (int)batch[nid];
    atomicAdd(&sums[g * CD + c], h[t]);
    if (c == 0) atomicAdd(&cnts[g], 1.0f);
}

__global__ void gt_final_kernel(const float* __restrict__ sums,
                                const float* __restrict__ cnts,
                                const float* __restrict__ Wlin,
                                const float* __restrict__ blin,
                                float* __restrict__ out) {
    int g = blockIdx.x;      // 0..63
    int j = threadIdx.x;     // 0..15
    float inv = 1.0f / fmaxf(cnts[g], 1.0f);
    float acc = blin[j];
#pragma unroll
    for (int c = 0; c < CD; ++c)
        acc += (sums[g * CD + c] * inv) * Wlin[c * 16 + j];
    out[g * 16 + j] = acc;
}

// ------------------------- host launcher -----------------------------------

extern "C" void kernel_launch(void* const* d_in, const int* in_sizes, int n_in,
                              void* d_out, int out_size, void* d_ws, size_t ws_size,
                              hipStream_t stream) {
    const float*     x     = (const float*)d_in[0];
    const long long* ei    = (const long long*)d_in[1];
    const long long* batch = (const long long*)d_in[2];
    const float *Wq1 = (const float*)d_in[3],  *bq1 = (const float*)d_in[4];
    const float *Wk1 = (const float*)d_in[5],  *bk1 = (const float*)d_in[6];
    const float *Wv1 = (const float*)d_in[7],  *bv1 = (const float*)d_in[8];
    const float *Ws1 = (const float*)d_in[9],  *bs1 = (const float*)d_in[10];
    const float *Wq2 = (const float*)d_in[11], *bq2 = (const float*)d_in[12];
    const float *Wk2 = (const float*)d_in[13], *bk2 = (const float*)d_in[14];
    const float *Wv2 = (const float*)d_in[15], *bv2 = (const float*)d_in[16];
    const float *Ws2 = (const float*)d_in[17], *bs2 = (const float*)d_in[18];
    const float *Wlin = (const float*)d_in[19], *blin = (const float*)d_in[20];

    const int N = in_sizes[0] / 128;   // 40000
    const int E = in_sizes[1] / 2;     // 640000
    const int G = 64;
    (void)n_in; (void)out_size; (void)ws_size;

    // ---- workspace carve-out (~117 MB) ----
    char* ws = (char*)d_ws;
    size_t off = 0;
    auto take = [&](size_t bytes) -> char* {
        char* p = ws + off;
        off = (off + bytes + 255) & ~(size_t)255;
        return p;
    };
    bf16_t*       xbf    = (bf16_t*)take((size_t)N * 128 * sizeof(bf16_t));
    float*        qkvs   = (float*) take((size_t)N * NODE_ROW * sizeof(float));
    bf16_t*       h1bf   = (bf16_t*)take((size_t)N * CD * sizeof(bf16_t));
    float*        h2f    = (float*) take((size_t)N * CD * sizeof(float));
    float*        alpha  = (float*) take((size_t)E * HD * sizeof(float));
    unsigned int* amax   = (unsigned int*)take((size_t)N * HD * 4);
    float*        denom  = (float*) take((size_t)N * HD * 4);
    float*        outacc = (float*) take((size_t)N * HD * CD * 4);
    bf16_t*       pW1    = (bf16_t*)take((size_t)4 * NTILES * 32 * 16 * 2);
    bf16_t*       pW2    = (bf16_t*)take((size_t)1 * NTILES * 32 * 16 * 2);
    float*        b1cat  = (float*) take(NODE_ROW * 4);
    float*        b2cat  = (float*) take(NODE_ROW * 4);
    float*        sums   = (float*) take((size_t)G * CD * 4);
    float*        cnts   = (float*) take((size_t)G * 4);

    const int TB = 256;
    dim3 gemm_blk(32, 4);
    dim3 gemm_grd(N / 16, (NTILES + 7) / 8);   // 8 N-tiles per block
    int edge_blocks = (E * HD + TB - 1) / TB;
    int node_blocks = (N * CD + TB - 1) / TB;

    // ---- prep ----
    gt_cast_bf16_kernel<<<(N * 128 + TB - 1) / TB, TB, 0, stream>>>(x, xbf, N * 128);
    gt_pack_bias_kernel<<<2, TB, 0, stream>>>(bq1, bk1, bv1, bs1, b1cat);
    gt_pack_bias_kernel<<<2, TB, 0, stream>>>(bq2, bk2, bv2, bs2, b2cat);
    gt_pack_w_kernel<<<(4 * NTILES * 32 + TB - 1) / TB, TB, 0, stream>>>(Wq1, Wk1, Wv1, Ws1, pW1, 128);
    gt_pack_w_kernel<<<(1 * NTILES * 32 + TB - 1) / TB, TB, 0, stream>>>(Wq2, Wk2, Wv2, Ws2, pW2, 32);

    // ================= layer 1 =================
    gt_wmma_gemm_kernel<128><<<gemm_grd, gemm_blk, 0, stream>>>(xbf, pW1, b1cat, qkvs);

    hipMemsetAsync(amax,   0, (size_t)N * HD * 4, stream);
    hipMemsetAsync(denom,  0, (size_t)N * HD * 4, stream);
    hipMemsetAsync(outacc, 0, (size_t)N * HD * CD * 4, stream);

    gt_edge_alpha_kernel      <<<edge_blocks, TB, 0, stream>>>(qkvs, ei, alpha, amax, E);
    gt_edge_exp_scatter_kernel<<<edge_blocks, TB, 0, stream>>>(qkvs, alpha, amax, denom, outacc, ei, E);
    gt_node_finish_kernel     <<<node_blocks, TB, 0, stream>>>(outacc, denom, qkvs, h1bf, h2f, 0, N);

    // ================= layer 2 =================
    gt_wmma_gemm_kernel<32><<<gemm_grd, gemm_blk, 0, stream>>>(h1bf, pW2, b2cat, qkvs);

    hipMemsetAsync(amax,   0, (size_t)N * HD * 4, stream);
    hipMemsetAsync(denom,  0, (size_t)N * HD * 4, stream);
    hipMemsetAsync(outacc, 0, (size_t)N * HD * CD * 4, stream);

    gt_edge_alpha_kernel      <<<edge_blocks, TB, 0, stream>>>(qkvs, ei, alpha, amax, E);
    gt_edge_exp_scatter_kernel<<<edge_blocks, TB, 0, stream>>>(qkvs, alpha, amax, denom, outacc, ei, E);
    gt_node_finish_kernel     <<<node_blocks, TB, 0, stream>>>(outacc, denom, qkvs, h1bf, h2f, 1, N);

    // ================= pool + head =================
    hipMemsetAsync(sums, 0, (size_t)G * CD * 4, stream);
    hipMemsetAsync(cnts, 0, (size_t)G * 4, stream);
    gt_pool_kernel <<<node_blocks, TB, 0, stream>>>(h2f, batch, sums, cnts, N);
    gt_final_kernel<<<G, 16, 0, stream>>>(sums, cnts, Wlin, blin, (float*)d_out);
}
